// MultiHeadAttention_18476949307735
// MI455X (gfx1250) — compile-verified, matrix operand order checked by reading, not run on hardware
//
#include <hip/hip_runtime.h>
#include <hip/hip_bf16.h>

#define D_MODEL 1024
#define HEADS   16
#define DK      64
#define SEQ     2048
#define BATCH   2
#define NROWS   (SEQ * BATCH)      /* 4096 flat rows, row = s*BATCH + b */
#define SM_SCALE 0.125f            /* 1/sqrt(64) */

typedef __bf16 bf16;
typedef __attribute__((ext_vector_type(16))) __bf16 v16bf;
typedef __attribute__((ext_vector_type(8)))  __bf16 v8bf;
typedef __attribute__((ext_vector_type(8)))  float  v8f;

static __device__ __forceinline__ bf16 f2bf(float f) {
  unsigned u = __builtin_bit_cast(unsigned, f);
  u += 0x7fffu + ((u >> 16) & 1u);                 // round-to-nearest-even
  return __builtin_bit_cast(bf16, (unsigned short)(u >> 16));
}

static __device__ __forceinline__ v8f vzero8() {
  v8f v;
  #pragma unroll
  for (int i = 0; i < 8; ++i) v[i] = 0.0f;
  return v;
}

#define WMMA_BF16(a, b, c) \
  __builtin_amdgcn_wmma_f32_16x16x32_bf16(false, (a), false, (b), (short)0, (c), false, false)

// CDNA5 async global->LDS copy (ASYNCcnt). Low 32 bits of a generic shared
// pointer are the wave-relative LDS byte offset (ISA 10.2 aperture rules).
#define ASYNC_B128(ldsptr, gptr)                                            \
  asm volatile("global_load_async_to_lds_b128 %0, %1, off"                  \
               :: "v"((unsigned)(size_t)(ldsptr)),                          \
                  "v"((unsigned long long)(gptr)) : "memory")
#define WAIT_ASYNC() asm volatile("s_wait_asynccnt 0x0" ::: "memory")

// ---------------------------------------------------------------------------
// fp32 -> bf16 conversion (4-wide)
// ---------------------------------------------------------------------------
__global__ void cvt_f32_to_bf16(const float* __restrict__ in,
                                bf16* __restrict__ out, int n4) {
  int i = blockIdx.x * blockDim.x + threadIdx.x;
  if (i < n4) {
    float4 f = ((const float4*)in)[i];
    union { bf16 b[4]; unsigned long long u; } r;
    r.b[0] = f2bf(f.x); r.b[1] = f2bf(f.y); r.b[2] = f2bf(f.z); r.b[3] = f2bf(f.w);
    ((unsigned long long*)out)[i] = r.u;
  }
}

// ---------------------------------------------------------------------------
// C[M,N] = A[M,K] * Bw[N,K]^T + bias ;  bf16 inputs, f32 accumulate.
// Block 128x128, 256 threads (8 waves), each wave 64x32 (4x2 WMMA tiles).
// Double-buffered LDS fed by async global->LDS copies.
// ---------------------------------------------------------------------------
#define GBM 128
#define GBN 128
#define GBK 32
#define GLD 40            /* LDS row stride (elems): 80B, 16B-aligned */
#define GBUF (GBM * GLD)  /* elems per buffer */

__launch_bounds__(256, 2)
__global__ void gemm_bf16(const bf16* __restrict__ A, const bf16* __restrict__ Bw,
                          const float* __restrict__ bias,
                          bf16* __restrict__ Cb, float* __restrict__ Cf,
                          int M, int N, int K) {
  __shared__ bf16 As[2 * GBUF];
  __shared__ bf16 Bs[2 * GBUF];

  const int tid  = threadIdx.x;
  const int lane = tid & 31;
  const int wave = tid >> 5;
  const int wm   = wave & 1;   // 0..1 : 64-row half of block
  const int wn   = wave >> 1;  // 0..3 : 32-col quarter of block
  const int lh   = lane >> 4;  // lane half (ISA fragment layout)
  const int lm   = lane & 15;
  const int mb   = blockIdx.x * GBM;
  const int nb   = blockIdx.y * GBN;

  // per-thread staging coords: 512 b128 chunks per matrix, 2 per thread each
  const int r0 = tid >> 2;                // rows 0..63
  const int r1 = r0 + 64;                 // rows 64..127
  const int cc = (tid & 3) * 8;           // col chunk

  auto stage = [&](int buf, int kb) {
    bf16* sA = As + buf * GBUF;
    bf16* sB = Bs + buf * GBUF;
    ASYNC_B128(&sA[r0 * GLD + cc], &A[(size_t)(mb + r0) * K + kb + cc]);
    ASYNC_B128(&sA[r1 * GLD + cc], &A[(size_t)(mb + r1) * K + kb + cc]);
    ASYNC_B128(&sB[r0 * GLD + cc], &Bw[(size_t)(nb + r0) * K + kb + cc]);
    ASYNC_B128(&sB[r1 * GLD + cc], &Bw[(size_t)(nb + r1) * K + kb + cc]);
  };

  v8f acc[4][2];
  #pragma unroll
  for (int i = 0; i < 4; ++i)
    #pragma unroll
    for (int j = 0; j < 2; ++j) acc[i][j] = vzero8();

  stage(0, 0);
  WAIT_ASYNC();
  __syncthreads();

  const int NKB = K / GBK;
  for (int t = 0; t < NKB; ++t) {
    const bf16* cA = As + (t & 1) * GBUF;
    const bf16* cB = Bs + (t & 1) * GBUF;
    if (t + 1 < NKB) stage((t + 1) & 1, (t + 1) * GBK);   // prefetch next slab

    // per-lane fragments: 16-bit A layout => K(e) = (e&7) + 8*g, g = half / 2+half
    v16bf af[4], bfr[2];
    #pragma unroll
    for (int i = 0; i < 4; ++i) {
      int r = wm * 64 + i * 16 + lm;
      *((v8bf*)&af[i])     = *(const v8bf*)&cA[r * GLD + 8 * lh];
      *((v8bf*)&af[i] + 1) = *(const v8bf*)&cA[r * GLD + 16 + 8 * lh];
    }
    #pragma unroll
    for (int j = 0; j < 2; ++j) {
      int r = wn * 32 + j * 16 + lm;
      *((v8bf*)&bfr[j])     = *(const v8bf*)&cB[r * GLD + 8 * lh];
      *((v8bf*)&bfr[j] + 1) = *(const v8bf*)&cB[r * GLD + 16 + 8 * lh];
    }
    #pragma unroll
    for (int i = 0; i < 4; ++i)
      #pragma unroll
      for (int j = 0; j < 2; ++j)
        acc[i][j] = WMMA_BF16(af[i], bfr[j], acc[i][j]);

    WAIT_ASYNC();        // next slab resident in other buffer
    __syncthreads();     // all waves done reading cur + see next slab
  }

  // epilogue: C layout -> (row m = 8*lh + r, col n = lm)
  #pragma unroll
  for (int i = 0; i < 4; ++i)
    #pragma unroll
    for (int j = 0; j < 2; ++j) {
      int ncol = nb + wn * 32 + j * 16 + lm;
      float bv = bias ? bias[ncol] : 0.0f;
      #pragma unroll
      for (int r = 0; r < 8; ++r) {
        int mrow = mb + wm * 64 + i * 16 + lh * 8 + r;
        float val = acc[i][j][r] + bv;
        if (Cf) Cf[(size_t)mrow * N + ncol] = val;
        else    Cb[(size_t)mrow * N + ncol] = f2bf(val);
      }
    }
}

// ---------------------------------------------------------------------------
// Flash attention: one block per (64 q rows, batch, head). 4 waves x 16 rows.
// ---------------------------------------------------------------------------
#define ABQ 64
#define ABK 32
#define KLD 72   /* [key 0..31][d 0..63] row stride (elems) */
#define VLD 40   /* [d 0..63][key 0..31] row stride          */
#define PLD 40   /* per-wave P tile [16][32] row stride       */

__launch_bounds__(128, 2)
__global__ void flash_attn(const bf16* __restrict__ Q, const bf16* __restrict__ Km,
                           const bf16* __restrict__ Vm, bf16* __restrict__ Ctx) {
  __shared__ bf16 Ks[ABK * KLD];
  __shared__ bf16 Vts[DK * VLD];
  __shared__ bf16 Ps[4 * 16 * PLD];

  const int tid  = threadIdx.x;
  const int lane = tid & 31;
  const int wave = tid >> 5;
  const int lh   = lane >> 4, lm = lane & 15;
  const int qblk = blockIdx.x;
  const int b    = blockIdx.y & (BATCH - 1);
  const int h    = blockIdx.y >> 1;
  const size_t headoff = (size_t)h * DK;

  // Q fragments for this wave's 16 rows (held for the whole loop)
  v16bf qf[2];
  {
    int s = qblk * ABQ + wave * 16 + lm;
    const bf16* qrow = Q + ((size_t)s * BATCH + b) * D_MODEL + headoff;
    #pragma unroll
    for (int kc = 0; kc < 2; ++kc) {
      *((v8bf*)&qf[kc])     = *(const v8bf*)&qrow[kc * 32 + 8 * lh];
      *((v8bf*)&qf[kc] + 1) = *(const v8bf*)&qrow[kc * 32 + 16 + 8 * lh];
    }
  }

  v8f o[4];
  float mi[8], li[8];
  #pragma unroll
  for (int t = 0; t < 4; ++t) o[t] = vzero8();
  #pragma unroll
  for (int r = 0; r < 8; ++r) { mi[r] = -1e30f; li[r] = 0.0f; }

  for (int kb = 0; kb < SEQ; kb += ABK) {
    __syncthreads();
    // K block: async global->LDS (row major); V block: transpose via VGPRs
    #pragma unroll
    for (int i = 0; i < 2; ++i) {
      int c = tid + i * 128;            // 256 chunks of 8 bf16
      int krow = c >> 3, col = (c & 7) * 8;
      size_t grow = ((size_t)(kb + krow) * BATCH + b) * D_MODEL + headoff + col;
      ASYNC_B128(&Ks[krow * KLD + col], &Km[grow]);
      v8bf vv = *(const v8bf*)&Vm[grow];
      #pragma unroll
      for (int j = 0; j < 8; ++j) Vts[(col + j) * VLD + krow] = vv[j];
    }
    WAIT_ASYNC();
    __syncthreads();

    // scores S = (Q K^T) * scale : two 16-wide key tiles, K-depth 64
    v8f s[2];
    #pragma unroll
    for (int nt = 0; nt < 2; ++nt) {
      v8f c = vzero8();
      #pragma unroll
      for (int kc = 0; kc < 2; ++kc) {
        v16bf kf;
        int krow = nt * 16 + lm;
        *((v8bf*)&kf)     = *(const v8bf*)&Ks[krow * KLD + kc * 32 + 8 * lh];
        *((v8bf*)&kf + 1) = *(const v8bf*)&Ks[krow * KLD + kc * 32 + 16 + 8 * lh];
        c = WMMA_BF16(qf[kc], kf, c);
      }
      s[nt] = c;
    }

    // online softmax (per row m = 8*lh + r; reduce over keys = over 16 lanes)
    #pragma unroll
    for (int r = 0; r < 8; ++r) {
      float s0 = s[0][r] * SM_SCALE, s1 = s[1][r] * SM_SCALE;
      float mx = fmaxf(s0, s1);
      #pragma unroll
      for (int msk = 1; msk < 16; msk <<= 1) mx = fmaxf(mx, __shfl_xor(mx, msk, 32));
      float mnew  = fmaxf(mi[r], mx);
      float alpha = __expf(mi[r] - mnew);
      float p0 = __expf(s0 - mnew), p1 = __expf(s1 - mnew);
      float rs = p0 + p1;
      #pragma unroll
      for (int msk = 1; msk < 16; msk <<= 1) rs += __shfl_xor(rs, msk, 32);
      li[r] = li[r] * alpha + rs;
      mi[r] = mnew;
      #pragma unroll
      for (int t = 0; t < 4; ++t) o[t][r] *= alpha;
      int m = lh * 8 + r;
      Ps[(wave * 16 + m) * PLD + lm]      = f2bf(p0);
      Ps[(wave * 16 + m) * PLD + 16 + lm] = f2bf(p1);
    }
    __syncthreads();

    // O += P @ V : A = P (16x32), B = Vt tiles (32x16), 4 d-tiles
    v16bf pf;
    *((v8bf*)&pf)     = *(const v8bf*)&Ps[(wave * 16 + lm) * PLD + 8 * lh];
    *((v8bf*)&pf + 1) = *(const v8bf*)&Ps[(wave * 16 + lm) * PLD + 16 + 8 * lh];
    #pragma unroll
    for (int t = 0; t < 4; ++t) {
      v16bf vf;
      int drow = t * 16 + lm;
      *((v8bf*)&vf)     = *(const v8bf*)&Vts[drow * VLD + 8 * lh];
      *((v8bf*)&vf + 1) = *(const v8bf*)&Vts[drow * VLD + 16 + 8 * lh];
      o[t] = WMMA_BF16(pf, vf, o[t]);
    }
  }

  // normalize and store context (bf16)
  #pragma unroll
  for (int r = 0; r < 8; ++r) {
    float inv = 1.0f / li[r];
    int s = qblk * ABQ + wave * 16 + lh * 8 + r;
    bf16* crow = Ctx + ((size_t)s * BATCH + b) * D_MODEL + headoff;
    #pragma unroll
    for (int t = 0; t < 4; ++t) crow[t * 16 + lm] = f2bf(o[t][r] * inv);
  }
}

// ---------------------------------------------------------------------------
extern "C" void kernel_launch(void* const* d_in, const int* in_sizes, int n_in,
                              void* d_out, int out_size, void* d_ws, size_t ws_size,
                              hipStream_t stream) {
  (void)in_sizes; (void)n_in; (void)out_size; (void)ws_size;
  const float* q_in = (const float*)d_in[0];
  const float* k_in = (const float*)d_in[1];
  const float* v_in = (const float*)d_in[2];
  const float* w_q  = (const float*)d_in[3];
  const float* b_q  = (const float*)d_in[4];
  const float* w_k  = (const float*)d_in[5];
  const float* b_k  = (const float*)d_in[6];
  const float* w_v  = (const float*)d_in[7];
  const float* b_v  = (const float*)d_in[8];
  const float* w_o  = (const float*)d_in[9];
  const float* b_o  = (const float*)d_in[10];

  const size_t R = (size_t)NROWS * D_MODEL;     // 4 Mi elems
  const size_t W = (size_t)D_MODEL * D_MODEL;   // 1 Mi elems
  bf16* ws  = (bf16*)d_ws;
  bf16* Qin = ws;           bf16* Kin = Qin + R;  bf16* Vin = Kin + R;
  bf16* Wq  = Vin + R;      bf16* Wk  = Wq + W;   bf16* Wv  = Wk + W;  bf16* Wo = Wv + W;
  bf16* Qp  = Wo + W;       bf16* Kp  = Qp + R;   bf16* Vp  = Kp + R;  bf16* Cx = Vp + R;

  // 1) fp32 -> bf16 conversions
  const float* srcs[7] = {q_in, k_in, v_in, w_q, w_k, w_v, w_o};
  bf16*        dsts[7] = {Qin,  Kin,  Vin,  Wq,  Wk,  Wv,  Wo};
  const size_t cnts[7] = {R, R, R, W, W, W, W};
  for (int i = 0; i < 7; ++i) {
    int n4 = (int)(cnts[i] / 4);
    cvt_f32_to_bf16<<<(n4 + 255) / 256, 256, 0, stream>>>(srcs[i], dsts[i], n4);
  }

  // 2) Q/K/V projections (bf16 out)
  dim3 gg(NROWS / GBM, D_MODEL / GBN);
  gemm_bf16<<<gg, 256, 0, stream>>>(Qin, Wq, b_q, Qp, nullptr, NROWS, D_MODEL, D_MODEL);
  gemm_bf16<<<gg, 256, 0, stream>>>(Kin, Wk, b_k, Kp, nullptr, NROWS, D_MODEL, D_MODEL);
  gemm_bf16<<<gg, 256, 0, stream>>>(Vin, Wv, b_v, Vp, nullptr, NROWS, D_MODEL, D_MODEL);

  // 3) attention -> context (bf16)
  dim3 ga(SEQ / ABQ, BATCH * HEADS);
  flash_attn<<<ga, 128, 0, stream>>>(Qp, Kp, Vp, Cx);

  // 4) output projection (fp32 out)
  gemm_bf16<<<gg, 256, 0, stream>>>(Cx, Wo, b_o, nullptr, (float*)d_out,
                                    NROWS, D_MODEL, D_MODEL);
}
